// Graph2text_69595650064838
// MI455X (gfx1250) — matrix operand order repaired and checked by reading, not compiled, not run on hardware
//
#include <hip/hip_runtime.h>
#include <hip/hip_bf16.h>

#define B_SZ   4
#define HH     8
#define DM     512
#define DK     64
#define LQ     1024
#define LK     2048
#define EPS_LN 1e-5f

typedef __attribute__((ext_vector_type(16))) __bf16 v16bf;
typedef __attribute__((ext_vector_type(8)))  __bf16 v8bf;
typedef __attribute__((ext_vector_type(8)))  float  v8f;

__device__ __forceinline__ v8f wmma_bf16(v16bf a, v16bf b, v8f c) {
  return __builtin_amdgcn_wmma_f32_16x16x32_bf16(
      /*neg_a=*/false, a, /*neg_b=*/false, b,
      /*c_mod=*/(short)0, c, /*reuse_a=*/false, /*reuse_b=*/false);
}

// ---- fragment loaders -------------------------------------------------------
// A 16x32 bf16: lane group g=lane>>4: elems 0..7 -> K=g*8+0..7 ; 8..15 -> K=16+g*8+0..7
__device__ __forceinline__ v16bf ldA_bf16(const __bf16* M, long ld, int row0, int k0) {
  int lane = threadIdx.x & 31;
  const __bf16* p = M + (long)(row0 + (lane & 15)) * ld + k0 + ((lane >> 4) << 3);
  v16bf a;
#pragma unroll
  for (int i = 0; i < 8; ++i) { a[i] = p[i]; a[i + 8] = p[i + 16]; }
  return a;
}
// B 32x16 bf16 from an LDS tile stored as [col][k] row-major: lane = column (lane&15),
// elems 0..7 -> K = (lane>>4)*8 + 0..7 ; 8..15 -> K = 16 + (lane>>4)*8 + 0..7
__device__ __forceinline__ v16bf ldB_lds(const __bf16* tile, int stride, int col0) {
  int lane = threadIdx.x & 31;
  const __bf16* p = tile + (col0 + (lane & 15)) * stride + ((lane >> 4) << 3);
  v16bf b;
#pragma unroll
  for (int i = 0; i < 8; ++i) { b[i] = p[i]; b[i + 8] = p[i + 16]; }
  return b;
}

// ---- async global->LDS tile staging (4KB tiles, 256 threads x 16B) ---------
// Flat shared address low 32 bits == LDS byte offset (LDS aperture in addr[63:32]).
__device__ __forceinline__ void stage64x32(__bf16* dst, const __bf16* src, long stride) {
  int t = threadIdx.x;
  const __bf16* g = src + (long)(t >> 2) * stride + ((t & 3) << 3);
  unsigned lds = (unsigned)(unsigned long long)dst + ((unsigned)t << 4);
  asm volatile("global_load_async_to_lds_b128 %0, %1, off" :: "v"(lds), "v"(g) : "memory");
}
__device__ __forceinline__ void stage32x64(__bf16* dst, const __bf16* src, long stride) {
  int t = threadIdx.x;
  const __bf16* g = src + (long)(t >> 3) * stride + ((t & 7) << 3);
  unsigned lds = (unsigned)(unsigned long long)dst + ((unsigned)t << 4);
  asm volatile("global_load_async_to_lds_b128 %0, %1, off" :: "v"(lds), "v"(g) : "memory");
}

// ---- kernel 0: f32 -> bf16 conversion (activations + weights, done once) ---
__global__ __launch_bounds__(256)
void cvt_bf16_kernel(const float* __restrict__ in, __bf16* __restrict__ out, long n) {
  long i = ((long)blockIdx.x * 256 + threadIdx.x) * 8;
  if (i >= n) return;
  v8bf o;
#pragma unroll
  for (int j = 0; j < 8; ++j) o[j] = (__bf16)in[i + j];
  *(v8bf*)(out + i) = o;
}

// ---- kernel 1: projections  out = X @ W^T + bias (pure bf16 GEMM) ----------
// transposeV==0 : out[((b*H+h)*L + r)*64 + d]
// transposeV==1 : out[((b*H+h)*64 + d)*L + r]   (V stored transposed for P@V)
__global__ __launch_bounds__(256)
void proj_kernel(const __bf16* __restrict__ X, const __bf16* __restrict__ W,
                 const float* __restrict__ bias, __bf16* __restrict__ out,
                 int L, int transposeV) {
  __shared__ __bf16 wt[2][64 * 32];   // double-buffered shared B tile

  int wave = threadIdx.x >> 5;
  int lane = threadIdx.x & 31;
  int g = lane >> 4, n = lane & 15;
  int m0 = blockIdx.x * 128 + wave * 16;
  int n0 = blockIdx.y * 64;

  const __bf16* Wt = W + (long)n0 * DM;
  stage64x32(wt[0], Wt, DM);          // prologue: K-step 0

  v8f acc[4] = {v8f{}, v8f{}, v8f{}, v8f{}};
#pragma unroll 1
  for (int s = 0; s < DM / 32; ++s) {
    __syncthreads();                  // all waves done reading buffer we overwrite
    if (s + 1 < DM / 32) {
      stage64x32(wt[(s + 1) & 1], Wt + (s + 1) * 32, DM);
      asm volatile("s_wait_asynccnt 0x1" ::: "memory");  // current tile landed
    } else {
      asm volatile("s_wait_asynccnt 0x0" ::: "memory");
    }
    __syncthreads();                  // every wave's copies for current tile done
    const __bf16* T = wt[s & 1];
    v16bf va = ldA_bf16(X, DM, m0, s * 32);
#pragma unroll
    for (int j = 0; j < 4; ++j) {
      v16bf vb = ldB_lds(T, 32, j * 16);
      acc[j] = wmma_bf16(va, vb, acc[j]);
    }
  }
#pragma unroll
  for (int j = 0; j < 4; ++j) {
    int e = n0 + j * 16 + n;
    float be = bias[e];
    int h = e >> 6, d = e & 63;
#pragma unroll
    for (int i = 0; i < 8; ++i) {
      int row = m0 + i + 8 * g;
      int b = row / L, r = row % L;
      float val = acc[j][i] + be;
      if (!transposeV)
        out[(((long)b * HH + h) * L + r) * DK + d] = (__bf16)val;
      else
        out[(((long)b * HH + h) * DK + d) * (long)L + r] = (__bf16)val;
    }
  }
}

// ---- kernel 2: flash attention (K/V tiles async-staged in LDS) -------------
#define PSTR 34
__global__ __launch_bounds__(256)
void attn_kernel(const __bf16* __restrict__ qb, const __bf16* __restrict__ kb,
                 const __bf16* __restrict__ vtg, const int* __restrict__ mask,
                 __bf16* __restrict__ attn_out) {
  __shared__ float  maskb[LK];                 // 8 KB additive mask bias
  __shared__ __bf16 pbuf[8 * 16 * PSTR];       // per-wave P-tile relayout
  __shared__ __bf16 smk[2][32 * 64];           // K tile:  32 keys x 64 d
  __shared__ __bf16 smv[2][64 * 32];           // V^T tile: 64 d x 32 keys

  int bh = blockIdx.y;
  int b = bh / HH, h = bh % HH;
  int wave = threadIdx.x >> 5;
  int lane = threadIdx.x & 31;
  int g = lane >> 4, n = lane & 15;
  int m0 = blockIdx.x * 128 + wave * 16;

  const int* mrow = mask + (long)b * LK;
  for (int i = threadIdx.x; i < LK; i += 256)
    maskb[i] = (mrow[i] == 0) ? -1e30f : 0.0f;

  const __bf16* qh = qb  + (long)bh * LQ * DK;
  const __bf16* kh = kb  + (long)bh * LK * DK;
  const __bf16* vh = vtg + (long)bh * DK * LK;

  v16bf qa0 = ldA_bf16(qh, DK, m0, 0);
  v16bf qa1 = ldA_bf16(qh, DK, m0, 32);

  float mrun[8], lsum[8];
#pragma unroll
  for (int i = 0; i < 8; ++i) { mrun[i] = -1e30f; lsum[i] = 0.0f; }
  v8f acc[4] = {v8f{}, v8f{}, v8f{}, v8f{}};

  __bf16* pw = pbuf + wave * 16 * PSTR;

  stage32x64(smk[0], kh, DK);                  // prologue: chunk 0
  stage64x32(smv[0], vh, LK);

#pragma unroll 1
  for (int step = 0; step < LK / 32; ++step) {
    int kc = step * 32;
    __syncthreads();
    if (step + 1 < LK / 32) {
      stage32x64(smk[(step + 1) & 1], kh + (long)(kc + 32) * DK, DK);
      stage64x32(smv[(step + 1) & 1], vh + (kc + 32), LK);
      asm volatile("s_wait_asynccnt 0x2" ::: "memory");
    } else {
      asm volatile("s_wait_asynccnt 0x0" ::: "memory");
    }
    __syncthreads();
    const __bf16* KT = smk[step & 1];
    const __bf16* VT = smv[step & 1];

    // scores 16x32 = q(16x64) @ k^T, operands from LDS
    v8f s0 = v8f{}, s1 = v8f{};
    s0 = wmma_bf16(qa0, ldB_lds(KT,      DK, 0),  s0);
    s0 = wmma_bf16(qa1, ldB_lds(KT + 32, DK, 0),  s0);
    s1 = wmma_bf16(qa0, ldB_lds(KT,      DK, 16), s1);
    s1 = wmma_bf16(qa1, ldB_lds(KT + 32, DK, 16), s1);

    float bias0 = maskb[kc + n];
    float bias1 = maskb[kc + 16 + n];
#pragma unroll
    for (int i = 0; i < 8; ++i) {
      s0[i] = s0[i] * 0.125f + bias0;          // 1/sqrt(64)
      s1[i] = s1[i] * 0.125f + bias1;
    }
    // online softmax; rows live across a 16-lane half (xor 1/2/4/8 stays inside)
#pragma unroll
    for (int i = 0; i < 8; ++i) {
      float c = fmaxf(s0[i], s1[i]);
#pragma unroll
      for (int off = 8; off > 0; off >>= 1) c = fmaxf(c, __shfl_xor(c, off, 32));
      float mn = fmaxf(mrun[i], c);
      float alpha = __expf(mrun[i] - mn);
      float p0 = __expf(s0[i] - mn);
      float p1 = __expf(s1[i] - mn);
      float rs = p0 + p1;
#pragma unroll
      for (int off = 8; off > 0; off >>= 1) rs += __shfl_xor(rs, off, 32);
      lsum[i] = lsum[i] * alpha + rs;
      mrun[i] = mn;
#pragma unroll
      for (int t = 0; t < 4; ++t) acc[t][i] *= alpha;
      int rloc = i + 8 * g;
      pw[rloc * PSTR + n]      = (__bf16)p0;   // C-layout -> LDS row-major
      pw[rloc * PSTR + 16 + n] = (__bf16)p1;
    }
    // reload P as A-fragment (per-wave region; DScnt orders the RAW)
    v16bf pa;
    {
      const __bf16* p = pw + (lane & 15) * PSTR + ((lane >> 4) << 3);
#pragma unroll
      for (int i = 0; i < 8; ++i) { pa[i] = p[i]; pa[i + 8] = p[i + 16]; }
    }
    // out += P(16x32) @ V(32x64) via LDS-resident V^T
#pragma unroll
    for (int t = 0; t < 4; ++t)
      acc[t] = wmma_bf16(pa, ldB_lds(VT, 32, t * 16), acc[t]);
  }

  float rinv[8];
#pragma unroll
  for (int i = 0; i < 8; ++i) rinv[i] = 1.0f / lsum[i];
#pragma unroll
  for (int t = 0; t < 4; ++t) {
    int d = t * 16 + n;
#pragma unroll
    for (int i = 0; i < 8; ++i) {
      int qrow = m0 + i + 8 * g;
      attn_out[((long)b * LQ + qrow) * DM + h * DK + d] = (__bf16)(acc[t][i] * rinv[i]);
    }
  }
}

// ---- kernel 3: output projection + bias + residual (bf16 GEMM, f32 out) ----
__global__ __launch_bounds__(256)
void oproj_kernel(const __bf16* __restrict__ A, const __bf16* __restrict__ Wo,
                  const float* __restrict__ bo, const float* __restrict__ Qin,
                  float* __restrict__ y) {
  __shared__ __bf16 wt[2][64 * 32];

  int wave = threadIdx.x >> 5;
  int lane = threadIdx.x & 31;
  int g = lane >> 4, n = lane & 15;
  int m0 = blockIdx.x * 128 + wave * 16;
  int n0 = blockIdx.y * 64;

  const __bf16* Wt = Wo + (long)n0 * DM;
  stage64x32(wt[0], Wt, DM);

  v8f acc[4] = {v8f{}, v8f{}, v8f{}, v8f{}};
#pragma unroll 1
  for (int s = 0; s < DM / 32; ++s) {
    __syncthreads();
    if (s + 1 < DM / 32) {
      stage64x32(wt[(s + 1) & 1], Wt + (s + 1) * 32, DM);
      asm volatile("s_wait_asynccnt 0x1" ::: "memory");
    } else {
      asm volatile("s_wait_asynccnt 0x0" ::: "memory");
    }
    __syncthreads();
    const __bf16* T = wt[s & 1];
    v16bf va = ldA_bf16(A, DM, m0, s * 32);
#pragma unroll
    for (int j = 0; j < 4; ++j)
      acc[j] = wmma_bf16(va, ldB_lds(T, 32, j * 16), acc[j]);
  }
#pragma unroll
  for (int j = 0; j < 4; ++j) {
    int e = n0 + j * 16 + n;
    float be = bo[e];
#pragma unroll
    for (int i = 0; i < 8; ++i) {
      long row = m0 + i + 8 * g;
      y[row * DM + e] = acc[j][i] + be + Qin[row * DM + e];
    }
  }
}

// ---- kernel 4: LayerNorm (one wave per 512-wide row) -----------------------
__global__ __launch_bounds__(256)
void ln_kernel(const float* __restrict__ y, const float* __restrict__ gamma,
               const float* __restrict__ beta, float* __restrict__ out) {
  int wave = threadIdx.x >> 5;
  int lane = threadIdx.x & 31;
  long row = (long)blockIdx.x * 8 + wave;
  const float* yr = y + row * DM;

  float v[16], s = 0.0f, ss = 0.0f;
#pragma unroll
  for (int j = 0; j < 16; ++j) {
    v[j] = yr[lane + j * 32];
    s += v[j];
    ss += v[j] * v[j];
  }
#pragma unroll
  for (int off = 16; off > 0; off >>= 1) {
    s  += __shfl_xor(s, off, 32);
    ss += __shfl_xor(ss, off, 32);
  }
  float mu = s * (1.0f / DM);
  float var = ss * (1.0f / DM) - mu * mu;
  float rstd = rsqrtf(var + EPS_LN);
  float* orow = out + row * DM;
#pragma unroll
  for (int j = 0; j < 16; ++j) {
    int c = lane + j * 32;
    orow[c] = (v[j] - mu) * rstd * gamma[c] + beta[c];
  }
}

// ---- host launch ------------------------------------------------------------
static inline int cvt_blocks(long n) { return (int)((n / 8 + 255) / 256); }

extern "C" void kernel_launch(void* const* d_in, const int* in_sizes, int n_in,
                              void* d_out, int out_size, void* d_ws, size_t ws_size,
                              hipStream_t stream) {
  const float* Q    = (const float*)d_in[0];
  const float* K    = (const float*)d_in[1];
  const float* V    = (const float*)d_in[2];
  const int*   mask = (const int*)d_in[4];
  const float* Wq   = (const float*)d_in[5];
  const float* bq   = (const float*)d_in[6];
  const float* Wk   = (const float*)d_in[7];
  const float* bk   = (const float*)d_in[8];
  const float* Wv   = (const float*)d_in[9];
  const float* bv   = (const float*)d_in[10];
  const float* Wo   = (const float*)d_in[11];
  const float* bo   = (const float*)d_in[12];
  const float* gam  = (const float*)d_in[13];
  const float* bet  = (const float*)d_in[14];

  const long NQ = (long)B_SZ * LQ * DM;   // 2M
  const long NK = (long)B_SZ * LK * DM;   // 4M
  const long NW = (long)DM * DM;          // 256K

  __bf16* Qx  = (__bf16*)d_ws;
  __bf16* Kx  = Qx  + NQ;
  __bf16* Vx  = Kx  + NK;
  __bf16* Wqb = Vx  + NK;
  __bf16* Wkb = Wqb + NW;
  __bf16* Wvb = Wkb + NW;
  __bf16* Wob = Wvb + NW;
  __bf16* qb  = Wob + NW;
  __bf16* kb  = qb  + (long)B_SZ * HH * LQ * DK;
  __bf16* vt  = kb  + (long)B_SZ * HH * LK * DK;
  __bf16* at  = vt  + (long)B_SZ * HH * LK * DK;
  float*  yb  = (float*)(at + NQ);

  cvt_bf16_kernel<<<cvt_blocks(NQ), 256, 0, stream>>>(Q,  Qx,  NQ);
  cvt_bf16_kernel<<<cvt_blocks(NK), 256, 0, stream>>>(K,  Kx,  NK);
  cvt_bf16_kernel<<<cvt_blocks(NK), 256, 0, stream>>>(V,  Vx,  NK);
  cvt_bf16_kernel<<<cvt_blocks(NW), 256, 0, stream>>>(Wq, Wqb, NW);
  cvt_bf16_kernel<<<cvt_blocks(NW), 256, 0, stream>>>(Wk, Wkb, NW);
  cvt_bf16_kernel<<<cvt_blocks(NW), 256, 0, stream>>>(Wv, Wvb, NW);
  cvt_bf16_kernel<<<cvt_blocks(NW), 256, 0, stream>>>(Wo, Wob, NW);

  proj_kernel<<<dim3((B_SZ * LQ) / 128, DM / 64), 256, 0, stream>>>(Qx, Wqb, bq, qb, LQ, 0);
  proj_kernel<<<dim3((B_SZ * LK) / 128, DM / 64), 256, 0, stream>>>(Kx, Wkb, bk, kb, LK, 0);
  proj_kernel<<<dim3((B_SZ * LK) / 128, DM / 64), 256, 0, stream>>>(Vx, Wvb, bv, vt, LK, 1);
  attn_kernel<<<dim3(LQ / 128, B_SZ * HH), 256, 0, stream>>>(qb, kb, vt, mask, at);
  oproj_kernel<<<dim3((B_SZ * LQ) / 128, DM / 64), 256, 0, stream>>>(at, Wob, bo, Q, yb);
  ln_kernel<<<dim3((B_SZ * LQ) / 8), 256, 0, stream>>>(yb, gam, bet, (float*)d_out);
}
